// DSAIndexer_13829794693311
// MI455X (gfx1250) — matrix operand order repaired and checked by reading, not compiled, not run on hardware
//
#include <hip/hip_runtime.h>
#include <hip/hip_bf16.h>
#include <stdint.h>

// ---- problem constants (B = 1) ----
#define SEQ     2048
#define NH      32
#define HD      128
#define RR      64
#define HIDDEN  4096
#define QLORA   1536
#define TOPK    2048

typedef __bf16 bf16_t;
typedef __attribute__((ext_vector_type(16))) __bf16 v16bf;
typedef __attribute__((ext_vector_type(8)))  __bf16 v8bf;
typedef __attribute__((ext_vector_type(8)))  float  v8f;

static_assert(sizeof(v8bf) == 16, "v8bf must be 16B");

// ---------------------------------------------------------------------------
// fragment helpers (wave32; A in ISA 7.12.2 table k order, B in contiguous k order)
// A (16x32, MxK): lane l -> row M = l&15 ; elems 0..7 = K[(l>>4)*8 .. +8),
//                 elems 8..15 = K[16+(l>>4)*8 .. +8)   (two contiguous 16B chunks)
// B (32x16, KxN): lane l -> col N = l&15 ; elems 0..15 = K[(l>>4)*16 .. +16)
//                 (one contiguous 32B chunk per lane in NT storage)
// ---------------------------------------------------------------------------
__device__ __forceinline__ v16bf load_a_frag(const bf16_t* __restrict__ rowbase,
                                             int ld, int lane) {
  const bf16_t* p = rowbase + (size_t)(lane & 15) * ld;
  int klo = (lane >> 4) * 8;
  v8bf lo = *(const v8bf*)(p + klo);
  v8bf hi = *(const v8bf*)(p + 16 + klo);
  return __builtin_shufflevector(lo, hi, 0,1,2,3,4,5,6,7,8,9,10,11,12,13,14,15);
}

__device__ __forceinline__ v16bf load_b_frag(const bf16_t* __restrict__ colbase,
                                             int ld, int lane) {
  const bf16_t* p = colbase + (size_t)(lane & 15) * ld + (lane >> 4) * 16;
  v8bf lo = *(const v8bf*)(p);
  v8bf hi = *(const v8bf*)(p + 8);
  return __builtin_shufflevector(lo, hi, 0,1,2,3,4,5,6,7,8,9,10,11,12,13,14,15);
}

__device__ __forceinline__ v8f wmma_bf16(v16bf a, v16bf b, v8f c) {
  return __builtin_amdgcn_wmma_f32_16x16x32_bf16(false, a, false, b,
                                                 (short)0, c, false, false);
}

// ---------------------------------------------------------------------------
// elementwise: f32 -> bf16 cast
// ---------------------------------------------------------------------------
__global__ void cast_f32_bf16(const float* __restrict__ in,
                              bf16_t* __restrict__ out, int n) {
  int i = blockIdx.x * 256 + threadIdx.x;
  if (i < n) out[i] = (bf16_t)in[i];
}

// transpose + cast: in (Rk x Cc) f32 row-major  ->  out (Cc x Rk) bf16 row-major
__global__ void transpose_cast(const float* __restrict__ in,
                               bf16_t* __restrict__ out, int Rk, int Cc) {
  __shared__ float tile[16][17];
  int r = blockIdx.y * 16 + threadIdx.y;
  int c = blockIdx.x * 16 + threadIdx.x;
  tile[threadIdx.y][threadIdx.x] = (r < Rk && c < Cc) ? in[(size_t)r * Cc + c] : 0.f;
  __syncthreads();
  int orow = blockIdx.x * 16 + threadIdx.y;  // = c dimension
  int ocol = blockIdx.y * 16 + threadIdx.x;  // = r dimension
  if (orow < Cc && ocol < Rk)
    out[(size_t)orow * Rk + ocol] = (bf16_t)tile[threadIdx.x][threadIdx.y];
}

// ---------------------------------------------------------------------------
// generic NT GEMM: C(MxN) = A(MxK, row-major bf16) * B(NxK, row-major bf16)^T
// block = 256 threads = 8 waves arranged 2(M) x 4(N); wave tile 16x64.
// ---------------------------------------------------------------------------
template <typename OutT>
__global__ __launch_bounds__(256) void gemm_nt(const bf16_t* __restrict__ A,
                                               const bf16_t* __restrict__ B,
                                               OutT* __restrict__ C,
                                               int M, int N, int K) {
  int wave  = threadIdx.x >> 5;
  int lane  = threadIdx.x & 31;
  int mBase = blockIdx.y * 32 + (wave >> 2) * 16;
  int nBase = blockIdx.x * 256 + (wave & 3) * 64;
  if (nBase >= N || mBase >= M) return;            // wave-uniform: EXEC stays all-ones

  int mRow = mBase + (lane & 15);
  if (mRow >= M) mRow = M - 1;                     // clamp loads; stores guarded
  const bf16_t* arow = A + (size_t)mRow * K;

  // clamp B rows per tile
  const bf16_t* brow[4];
#pragma unroll
  for (int t = 0; t < 4; ++t) {
    int n = nBase + t * 16 + (lane & 15);
    if (n >= N) n = N - 1;
    brow[t] = B + (size_t)n * K + (lane >> 4) * 16;
  }

  v8f acc[4] = {};
  int klo = (lane >> 4) * 16;
  for (int k = 0; k < K; k += 32) {
    const bf16_t* pa = arow + k;
    __builtin_prefetch(pa + 64, 0, 1);
    v8bf alo = *(const v8bf*)(pa + klo);
    v8bf ahi = *(const v8bf*)(pa + 8 + klo);
    v16bf af = __builtin_shufflevector(alo, ahi,
                 0,1,2,3,4,5,6,7,8,9,10,11,12,13,14,15);
#pragma unroll
    for (int t = 0; t < 4; ++t) {
      const bf16_t* pb = brow[t] + k;
      v8bf blo = *(const v8bf*)(pb);
      v8bf bhi = *(const v8bf*)(pb + 8);
      v16bf bfr = __builtin_shufflevector(blo, bhi,
                    0,1,2,3,4,5,6,7,8,9,10,11,12,13,14,15);
      acc[t] = wmma_bf16(af, bfr, acc[t]);
    }
  }

  // C layout: elem r of acc -> row mBase + (lane>>4)*8 + r, col nBase+t*16+(lane&15)
  int rowOff = mBase + (lane >> 4) * 8;
  int col0   = lane & 15;
#pragma unroll
  for (int t = 0; t < 4; ++t) {
    int n = nBase + t * 16 + col0;
    if (n >= N) continue;
#pragma unroll
    for (int r = 0; r < 8; ++r) {
      int m = rowOff + r;
      if (m < M) C[(size_t)m * N + n] = (OutT)acc[t][r];
    }
  }
}

// ---------------------------------------------------------------------------
// RoPE on q: qlin (SEQ x NH x HD bf16) -> q (same), rope on first RR dims/head
// ---------------------------------------------------------------------------
__global__ void rope_q(const bf16_t* __restrict__ qlin,
                       const float* __restrict__ cosp,
                       const float* __restrict__ sinp,
                       bf16_t* __restrict__ qout) {
  int i = blockIdx.x * 256 + threadIdx.x;          // over SEQ*NH*HD
  int d = i & (HD - 1);
  int s = i >> 12;                                  // i / (NH*HD)
  float x = (float)qlin[i];
  float o = x;
  if (d < RR) {
    float c  = cosp[s * RR + d];
    float sn = sinp[s * RR + d];
    float rot = (d < RR / 2) ? -(float)qlin[i + RR / 2] : (float)qlin[i - RR / 2];
    o = x * c + rot * sn;
  }
  qout[i] = (bf16_t)o;
}

// ---------------------------------------------------------------------------
// k epilogue: layernorm over HD then rope on first RR dims -> bf16
// one block (128 threads) per sequence position
// ---------------------------------------------------------------------------
__global__ __launch_bounds__(HD) void k_ln_rope(const float* __restrict__ kraw,
                                                const float* __restrict__ gamma,
                                                const float* __restrict__ beta,
                                                const float* __restrict__ cosp,
                                                const float* __restrict__ sinp,
                                                bf16_t* __restrict__ kout) {
  __shared__ float red[HD];
  __shared__ float v[HD];
  int s = blockIdx.x, d = threadIdx.x;
  float x = kraw[(size_t)s * HD + d];
  red[d] = x; __syncthreads();
  for (int off = HD / 2; off > 0; off >>= 1) {
    if (d < off) red[d] += red[d + off];
    __syncthreads();
  }
  float mu = red[0] * (1.f / HD);
  __syncthreads();
  float dx = x - mu;
  red[d] = dx * dx; __syncthreads();
  for (int off = HD / 2; off > 0; off >>= 1) {
    if (d < off) red[d] += red[d + off];
    __syncthreads();
  }
  float var = red[0] * (1.f / HD);
  float nv = dx * rsqrtf(var + 1e-6f) * gamma[d] + beta[d];
  v[d] = nv; __syncthreads();
  float o = nv;
  if (d < RR) {
    float rot = (d < RR / 2) ? -v[d + RR / 2] : v[d - RR / 2];
    o = nv * cosp[s * RR + d] + rot * sinp[s * RR + d];
  }
  kout[(size_t)s * HD + d] = (bf16_t)o;
}

// ---------------------------------------------------------------------------
// fused scores + head-combine:
//   index_scores[s,t] = sum_h w[s,h]*H^-0.5 * relu(D^-0.5 * q[s,h,:].k[t,:])
// block: 256 thr = 8 waves; each wave: (s_local = w>>1, head-half = w&1)
// k tile (TT x HD bf16) staged to LDS via CDNA5 async global->LDS loads.
// ---------------------------------------------------------------------------
#define TT 128
__global__ __launch_bounds__(256) void scores_combine(
    const bf16_t* __restrict__ q, const bf16_t* __restrict__ kmat,
    const float* __restrict__ wraw, float* __restrict__ iscore) {
  __shared__ __attribute__((aligned(16))) bf16_t ktile[TT * HD];   // 32 KB
  __shared__ float partial[4 * TT];                                // 2 KB

  int tid  = threadIdx.x;
  int lane = tid & 31;
  int wave = tid >> 5;
  int sBlk = blockIdx.y;            // 512 blocks of 4 s
  int tBlk = blockIdx.x;            // 16 blocks of TT t

  for (int i = tid; i < 4 * TT; i += 256) partial[i] = 0.f;

  // async stage k tile: TT*HD*2 bytes = 2048 x b128, 8 per thread
  {
    const bf16_t* src = kmat + (size_t)tBlk * TT * HD;
    for (int i = tid; i < (TT * HD) / 8; i += 256) {
      uint32_t la = (uint32_t)(uintptr_t)(&ktile[i * 8]);
      uint64_t ga = (uint64_t)(uintptr_t)(src + i * 8);
      asm volatile("global_load_async_to_lds_b128 %0, %1, off"
                   :: "v"(la), "v"(ga) : "memory");
    }
    asm volatile("s_wait_asynccnt 0" ::: "memory");
  }
  __syncthreads();

  int sLoc  = wave >> 1;
  int htile = wave & 1;
  int s = sBlk * 4 + sLoc;
  const bf16_t* abase = q + (size_t)s * (NH * HD) + (size_t)htile * 16 * HD;

  v8f acc[TT / 16] = {};
  int klo = (lane >> 4) * 16;
  const bf16_t* arow = abase + (size_t)(lane & 15) * HD;
  for (int k = 0; k < HD; k += 32) {
    v8bf alo = *(const v8bf*)(arow + k + klo);
    v8bf ahi = *(const v8bf*)(arow + k + 8 + klo);
    v16bf af = __builtin_shufflevector(alo, ahi,
                 0,1,2,3,4,5,6,7,8,9,10,11,12,13,14,15);
#pragma unroll
    for (int t = 0; t < TT / 16; ++t) {
      const bf16_t* pb = &ktile[(size_t)(t * 16 + (lane & 15)) * HD + k + (lane >> 4) * 16];
      v8bf blo = *(const v8bf*)(pb);            // ds_read_b128
      v8bf bhi = *(const v8bf*)(pb + 8);
      v16bf bfr = __builtin_shufflevector(blo, bhi,
                    0,1,2,3,4,5,6,7,8,9,10,11,12,13,14,15);
      acc[t] = wmma_bf16(af, bfr, acc[t]);
    }
  }

  // per-lane weights for the 8 head-rows this lane's acc VGPRs hold
  const float wscale = 0.17677669529663689f;      // 32^-0.5
  const float kscale = 0.08838834764831845f;      // 128^-0.5
  float wl[8];
#pragma unroll
  for (int r = 0; r < 8; ++r) {
    int h = htile * 16 + (lane >> 4) * 8 + r;
    wl[r] = wraw[(size_t)s * NH + h] * wscale;
  }

#pragma unroll
  for (int t = 0; t < TT / 16; ++t) {
    float p = 0.f;
#pragma unroll
    for (int r = 0; r < 8; ++r) {
      float sc = acc[t][r] * kscale;
      sc = sc > 0.f ? sc : 0.f;
      p += sc * wl[r];
    }
    p += __shfl_xor(p, 16, 32);                   // combine M rows 0-7 and 8-15
    if ((lane & 16) == 0)
      atomicAdd(&partial[sLoc * TT + t * 16 + (lane & 15)], p);  // ds_add_f32
  }
  __syncthreads();

  for (int i = tid; i < 4 * TT; i += 256) {
    int so = sBlk * 4 + i / TT;
    int to = tBlk * TT + (i & (TT - 1));
    iscore[(size_t)so * SEQ + to] = partial[i];
  }
}

// ---------------------------------------------------------------------------
// top_k(2048 of 2048) == full descending argsort per row (bitonic, in LDS)
// ---------------------------------------------------------------------------
__global__ __launch_bounds__(1024) void topk_sort(const float* __restrict__ iscore,
                                                  int* __restrict__ out) {
  __shared__ float key[SEQ];
  __shared__ int   idx[SEQ];
  int row = blockIdx.x, tid = threadIdx.x;
  for (int i = tid; i < SEQ; i += 1024) {
    key[i] = iscore[(size_t)row * SEQ + i];
    idx[i] = i;
  }
  __syncthreads();
  for (int k = 2; k <= SEQ; k <<= 1) {
    for (int j = k >> 1; j > 0; j >>= 1) {
      for (int e = tid; e < SEQ; e += 1024) {
        int p = e ^ j;
        if (p > e) {
          bool up = ((e & k) == 0);
          float ka = key[e], kb = key[p];
          int ia = idx[e], ib = idx[p];
          // "before" == a precedes b in final (descending key, ascending idx) order
          bool before = (ka > kb) || (ka == kb && ia < ib);
          if (up ? !before : before) {
            key[e] = kb; key[p] = ka;
            idx[e] = ib; idx[p] = ia;
          }
        }
      }
      __syncthreads();
    }
  }
  for (int i = tid; i < SEQ; i += 1024) out[(size_t)row * SEQ + i] = idx[i];
}

// ---------------------------------------------------------------------------
// host-side launch
// ---------------------------------------------------------------------------
extern "C" void kernel_launch(void* const* d_in, const int* in_sizes, int n_in,
                              void* d_out, int out_size, void* d_ws, size_t ws_size,
                              hipStream_t stream) {
  (void)in_sizes; (void)n_in; (void)out_size; (void)ws_size;
  const float* hidden = (const float*)d_in[0];   // (SEQ, HIDDEN)
  const float* qresid = (const float*)d_in[1];   // (SEQ, QLORA)
  const float* cosp   = (const float*)d_in[2];   // (SEQ, RR)
  const float* sinp   = (const float*)d_in[3];   // (SEQ, RR)
  const float* Wqb    = (const float*)d_in[4];   // (QLORA, NH*HD)
  const float* Wk     = (const float*)d_in[5];   // (HIDDEN, HD)
  const float* gamma  = (const float*)d_in[6];   // (HD)
  const float* beta   = (const float*)d_in[7];   // (HD)
  const float* Ww     = (const float*)d_in[8];   // (HIDDEN, NH)
  int* out = (int*)d_out;

  // workspace layout (bytes), total ~89 MB
  char* ws = (char*)d_ws;
  bf16_t* hid_bf  = (bf16_t*)(ws + 0);                       // 16.8 MB
  bf16_t* qr_bf   = (bf16_t*)(ws + 16777216);                //  6.3 MB
  bf16_t* WqbT    = (bf16_t*)(ws + 23068672);                // 12.6 MB (NHHD x QLORA)
  bf16_t* WkT     = (bf16_t*)(ws + 35651584);                //  1.0 MB (HD x HIDDEN)
  bf16_t* WwT     = (bf16_t*)(ws + 36700160);                //  0.25 MB (NH x HIDDEN)
  bf16_t* qlin_bf = (bf16_t*)(ws + 36962304);                // 16.8 MB
  bf16_t* q_bf    = (bf16_t*)(ws + 53739520);                // 16.8 MB
  float*  kraw    = (float*) (ws + 70516736);                //  1.0 MB
  float*  wraw    = (float*) (ws + 71565312);                //  0.25 MB
  bf16_t* k_bf    = (bf16_t*)(ws + 71827456);                //  0.5 MB
  float*  iscore  = (float*) (ws + 72351744);                // 16.8 MB

  // 1) precision conversion + weight transposes (NT layout for WMMA)
  cast_f32_bf16<<<(SEQ * HIDDEN) / 256, 256, 0, stream>>>(hidden, hid_bf, SEQ * HIDDEN);
  cast_f32_bf16<<<(SEQ * QLORA) / 256, 256, 0, stream>>>(qresid, qr_bf, SEQ * QLORA);
  dim3 tb(16, 16);
  transpose_cast<<<dim3((NH * HD) / 16, QLORA / 16),  tb, 0, stream>>>(Wqb, WqbT, QLORA, NH * HD);
  transpose_cast<<<dim3(HD / 16,        HIDDEN / 16), tb, 0, stream>>>(Wk,  WkT,  HIDDEN, HD);
  transpose_cast<<<dim3(NH / 16,        HIDDEN / 16), tb, 0, stream>>>(Ww,  WwT,  HIDDEN, NH);

  // 2) GEMMs (bf16 WMMA, f32 accumulate)
  gemm_nt<bf16_t><<<dim3((NH * HD) / 256, SEQ / 32), 256, 0, stream>>>(
      qr_bf, WqbT, qlin_bf, SEQ, NH * HD, QLORA);
  gemm_nt<float><<<dim3(1, SEQ / 32), 256, 0, stream>>>(
      hid_bf, WkT, kraw, SEQ, HD, HIDDEN);
  gemm_nt<float><<<dim3(1, SEQ / 32), 256, 0, stream>>>(
      hid_bf, WwT, wraw, SEQ, NH, HIDDEN);

  // 3) epilogues: RoPE(q), LN+RoPE(k)
  rope_q<<<(SEQ * NH * HD) / 256, 256, 0, stream>>>(qlin_bf, cosp, sinp, q_bf);
  k_ln_rope<<<SEQ, HD, 0, stream>>>(kraw, gamma, beta, cosp, sinp, k_bf);

  // 4) fused scoring + head combine (async-LDS k tiles + WMMA)
  scores_combine<<<dim3(SEQ / TT, SEQ / 4), 256, 0, stream>>>(q_bf, k_bf, wraw, iscore);

  // 5) top-k == full descending argsort per row
  topk_sort<<<SEQ, 1024, 0, stream>>>(iscore, out);
}
